// GAT_layer_5016521801746
// MI455X (gfx1250) — compile-verified, matrix operand order checked by reading, not run on hardware
//
#include <hip/hip_runtime.h>

#define NNODES 50000
#define NEDGES 800000
#define ET     850000      // edges + self loops
#define DIN    128
#define DOUT   64
#define NH     2
#define NLR    256         // columns of combined [x_l | x_r]
#define NEG_SLOPE 0.2f
#define LN_EPS 1e-5f

typedef __attribute__((ext_vector_type(16))) _Float16 v16h;
typedef __attribute__((ext_vector_type(8)))  _Float16 v8h;
typedef __attribute__((ext_vector_type(8)))  float    v8f;

// ---- workspace layout (bytes) ----
#define WS_XW     0u          // 50000*256 f32 = 51,200,000
#define WS_WLR    51200000u   // 32768 f16 (swizzled W_l|W_r)
#define WS_WF     51265536u   // 8192 f16 (swizzled W_f)
#define WS_AMAX   51281920u   // 100000 u32
#define WS_DEN    51681920u   // 100000 f32
#define WS_ALPHA  52081920u   // 1700000 f32
#define WS_AGG    58881920u   // 6400000 f32
// total ~84.5 MB

// Map (k-within-32, n-within-16) to the per-lane v16h operand slot used by
// v_wmma_f32_16x16x32_f16 (16-bit A/B layout, ISA 7.12.2).
__device__ __forceinline__ int swz_idx(int nt, int kstep, int kr, int nl) {
  int lane = nl + (((kr >> 3) & 1) << 4);           // K 8-15/24-31 -> lanes 16-31
  int i    = (kr & 7) + (((kr >> 4) & 1) << 3);     // K 16-31 -> elems 8-15
  return ((nt * 4 + kstep) * 32 + lane) * 16 + i;
}

// ---------------- weight conversion + swizzle ----------------
__global__ void k_swizzle(const float* __restrict__ Wl, const float* __restrict__ Wr,
                          const float* __restrict__ Wf,
                          _Float16* __restrict__ wlr, _Float16* __restrict__ wf) {
  int t = blockIdx.x * blockDim.x + threadIdx.x;
  if (t < 128 * 256) {
    int k = t >> 8, n = t & 255;
    float v = (n < 128) ? Wl[k * 128 + n] : Wr[k * 128 + (n - 128)];
    wlr[swz_idx(n >> 4, k >> 5, k & 31, n & 15)] = (_Float16)v;
  } else {
    int t2 = t - 128 * 256;
    if (t2 < 128 * 64) {
      int k = t2 >> 6, n = t2 & 63;
      wf[swz_idx(n >> 4, k >> 5, k & 31, n & 15)] = (_Float16)Wf[k * 64 + n];
    }
  }
}

// ---------------- fused x_l|x_r GEMM: XW[50000,256] = X @ [W_l|W_r] + [b_l|b_r] ----------------
__global__ __launch_bounds__(256) void k_gemm_xlr(const float* __restrict__ X,
                                                  const float* __restrict__ bl,
                                                  const float* __restrict__ br,
                                                  const _Float16* __restrict__ wlr,
                                                  float* __restrict__ XW) {
  __shared__ __attribute__((aligned(16))) _Float16 As[16][136];  // 16B-pad rows
  int tid = threadIdx.x, bx = blockIdx.x, by = blockIdx.y;
  {   // stage 16x128 A tile -> f16 LDS
    int row = tid >> 4, cb = tid & 15;
    const float* src = X + (size_t)(bx * 16 + row) * DIN + cb * 8;
    float4 f0 = ((const float4*)src)[0];
    float4 f1 = ((const float4*)src)[1];
    _Float16* d = &As[row][cb * 8];
    d[0]=(_Float16)f0.x; d[1]=(_Float16)f0.y; d[2]=(_Float16)f0.z; d[3]=(_Float16)f0.w;
    d[4]=(_Float16)f1.x; d[5]=(_Float16)f1.y; d[6]=(_Float16)f1.z; d[7]=(_Float16)f1.w;
  }
  __syncthreads();
  int wave = tid >> 5, lane = tid & 31;
  int nt = by * 8 + wave;                 // N tile 0..15
  int hl = (lane >> 4) & 1, mrow = lane & 15;
  v8f c = {};
#pragma unroll
  for (int ks = 0; ks < 4; ++ks) {
    int k0 = ks * 32;
    v8h alo = *(const v8h*)&As[mrow][k0 + hl * 8];
    v8h ahi = *(const v8h*)&As[mrow][k0 + 16 + hl * 8];
    v16h a = __builtin_shufflevector(alo, ahi, 0,1,2,3,4,5,6,7,8,9,10,11,12,13,14,15);
    v16h b = *(const v16h*)(wlr + ((nt * 4 + ks) * 32 + lane) * 16);
    c = __builtin_amdgcn_wmma_f32_16x16x32_f16(false, a, false, b, (short)0, c, false, false);
  }
  int n = nt * 16 + mrow;
  float bias = (n < 128) ? bl[n] : br[n - 128];
#pragma unroll
  for (int i = 0; i < 8; ++i) {
    int row = bx * 16 + i + hl * 8;       // C layout: lanes0-15 M=i, lanes16-31 M=i+8
    XW[(size_t)row * NLR + n] = c[i] + bias;
  }
}

// ---------------- init: agg = GAT bias (broadcast), den = 0, amax = encoded -inf ----------------
__global__ void k_init(const float* __restrict__ bias_g, float* __restrict__ agg,
                       float* __restrict__ den, unsigned* __restrict__ amax) {
  int i = blockIdx.x * blockDim.x + threadIdx.x;
  if (i < NNODES * DIN) agg[i] = bias_g[i & 127];
  if (i < NNODES * NH) { den[i] = 0.f; amax[i] = 0u; }
}

// ---------------- per-edge GATv2 score + atomic segment max (8 lanes per edge-head) ----------------
__global__ __launch_bounds__(256) void k_edge_score(const int* __restrict__ E,
                                                    const float* __restrict__ XW,
                                                    const float* __restrict__ att,
                                                    float* __restrict__ alpha,
                                                    unsigned* __restrict__ amax) {
  int gt = blockIdx.x * 256 + threadIdx.x;
  int g = gt >> 3, sub = gt & 7;
  if (g >= ET * NH) return;
  int e = g >> 1, h = g & 1;
  int s, d;
  if (e < NEDGES) { s = E[e]; d = E[NEDGES + e]; } else { s = e - NEDGES; d = s; }
  const float* pl = XW + (size_t)s * NLR + h * DOUT;
  const float* pr = XW + (size_t)d * NLR + 128 + h * DOUT;
  const float* pa = att + h * DOUT;
  float acc = 0.f;
#pragma unroll
  for (int k = 0; k < 8; ++k) {
    int ch = sub + k * 8;
    float v = pl[ch] + pr[ch];
    v = v > 0.f ? v : NEG_SLOPE * v;
    acc += pa[ch] * v;
  }
  acc += __shfl_xor(acc, 4, 8);
  acc += __shfl_xor(acc, 2, 8);
  acc += __shfl_xor(acc, 1, 8);
  if (sub == 0) {
    alpha[g] = acc;
    unsigned b = __float_as_uint(acc);
    b = (b & 0x80000000u) ? ~b : (b | 0x80000000u);   // monotone float->uint
    atomicMax(&amax[d * NH + h], b);
  }
}

// ---------------- exp(alpha - max) + atomic segment sum ----------------
__global__ void k_edge_expsum(const int* __restrict__ E, float* __restrict__ alpha,
                              const unsigned* __restrict__ amax, float* __restrict__ den) {
  int i = blockIdx.x * blockDim.x + threadIdx.x;
  if (i >= ET * NH) return;
  int e = i >> 1, h = i & 1;
  int d = (e < NEDGES) ? E[NEDGES + e] : e - NEDGES;
  unsigned u = amax[d * NH + h];
  float m = (u & 0x80000000u) ? __uint_as_float(u & 0x7fffffffu) : __uint_as_float(~u);
  float w = expf(alpha[i] - m);
  alpha[i] = w;
  unsafeAtomicAdd(&den[d * NH + h], w);
}

// ---------------- weighted scatter aggregate (one wave per edge) ----------------
__global__ __launch_bounds__(256) void k_aggregate(const int* __restrict__ E,
                                                   const float* __restrict__ XW,
                                                   const float* __restrict__ alpha,
                                                   const float* __restrict__ den,
                                                   float* __restrict__ agg) {
  int e = (blockIdx.x * 256 + threadIdx.x) >> 5;
  int lane = threadIdx.x & 31;
  if (e >= ET) return;
  int s, d;
  if (e < NEDGES) { s = E[e]; d = E[NEDGES + e]; } else { s = e - NEDGES; d = s; }
  float w0 = alpha[e * 2 + 0] / den[d * 2 + 0];
  float w1 = alpha[e * 2 + 1] / den[d * 2 + 1];
  const float* src = XW + (size_t)s * NLR;       // x_l lives in cols 0..127
  float* dst = agg + (size_t)d * DIN;
#pragma unroll
  for (int j = 0; j < 4; ++j) {
    int ch = lane + j * 32;
    float wgt = (ch < 64) ? w0 : w1;
    unsafeAtomicAdd(&dst[ch], src[ch] * wgt);
  }
}

// ---------------- final linear (WMMA) + fused LayerNorm ----------------
__global__ __launch_bounds__(128) void k_final_ln(const float* __restrict__ agg,
                                                  const _Float16* __restrict__ wf,
                                                  const float* __restrict__ bf,
                                                  const float* __restrict__ gamma,
                                                  const float* __restrict__ beta,
                                                  float* __restrict__ out) {
  __shared__ __attribute__((aligned(16))) _Float16 As[16][136];
  __shared__ float Hs[16][68];
  int tid = threadIdx.x, bx = blockIdx.x;
  {   // stage 16x128 A tile (agg already includes GAT bias)
    int row = tid >> 3, cb = tid & 7;
    const float* src = agg + (size_t)(bx * 16 + row) * DIN + cb * 16;
    _Float16* dp = &As[row][cb * 16];
#pragma unroll
    for (int q = 0; q < 4; ++q) {
      float4 f = ((const float4*)src)[q];
      dp[q*4+0]=(_Float16)f.x; dp[q*4+1]=(_Float16)f.y;
      dp[q*4+2]=(_Float16)f.z; dp[q*4+3]=(_Float16)f.w;
    }
  }
  __syncthreads();
  int wave = tid >> 5, lane = tid & 31;           // wave = N tile (0..3)
  int hl = (lane >> 4) & 1, mrow = lane & 15;
  v8f c = {};
#pragma unroll
  for (int ks = 0; ks < 4; ++ks) {
    int k0 = ks * 32;
    v8h alo = *(const v8h*)&As[mrow][k0 + hl * 8];
    v8h ahi = *(const v8h*)&As[mrow][k0 + 16 + hl * 8];
    v16h a = __builtin_shufflevector(alo, ahi, 0,1,2,3,4,5,6,7,8,9,10,11,12,13,14,15);
    v16h b = *(const v16h*)(wf + ((wave * 4 + ks) * 32 + lane) * 16);
    c = __builtin_amdgcn_wmma_f32_16x16x32_f16(false, a, false, b, (short)0, c, false, false);
  }
  int n = wave * 16 + mrow;
#pragma unroll
  for (int i = 0; i < 8; ++i) Hs[i + hl * 8][n] = c[i] + bf[n];
  __syncthreads();
  // LayerNorm over 64 cols; 8 threads per row
  int r = tid >> 3, sub = tid & 7;
  float s1 = 0.f, s2 = 0.f;
#pragma unroll
  for (int k = 0; k < 8; ++k) {
    float v = Hs[r][sub + k * 8];
    s1 += v; s2 += v * v;
  }
  s1 += __shfl_xor(s1, 4, 8); s2 += __shfl_xor(s2, 4, 8);
  s1 += __shfl_xor(s1, 2, 8); s2 += __shfl_xor(s2, 2, 8);
  s1 += __shfl_xor(s1, 1, 8); s2 += __shfl_xor(s2, 1, 8);
  float mu  = s1 * (1.f / 64.f);
  float var = s2 * (1.f / 64.f) - mu * mu;
  float inv = rsqrtf(var + LN_EPS);
  float* orow = out + (size_t)(bx * 16 + r) * DOUT;
#pragma unroll
  for (int k = 0; k < 8; ++k) {
    int ch = sub + k * 8;
    orow[ch] = (Hs[r][ch] - mu) * inv * gamma[ch] + beta[ch];
  }
}

extern "C" void kernel_launch(void* const* d_in, const int* in_sizes, int n_in,
                              void* d_out, int out_size, void* d_ws, size_t ws_size,
                              hipStream_t stream) {
  const float* X      = (const float*)d_in[0];
  const int*   E      = (const int*)  d_in[1];
  const float* attr   = (const float*)d_in[2];
  const float* Wl     = (const float*)d_in[3];
  const float* bl     = (const float*)d_in[4];
  const float* Wr     = (const float*)d_in[5];
  const float* br     = (const float*)d_in[6];
  const float* att    = (const float*)d_in[7];
  const float* bias_g = (const float*)d_in[8];
  const float* Wf     = (const float*)d_in[9];
  const float* bf     = (const float*)d_in[10];
  const float* gamma  = (const float*)d_in[11];
  const float* beta   = (const float*)d_in[12];

  char* ws = (char*)d_ws;
  float*     XW    = (float*)(ws + WS_XW);
  _Float16*  wlr   = (_Float16*)(ws + WS_WLR);
  _Float16*  wfz   = (_Float16*)(ws + WS_WF);
  unsigned*  amax  = (unsigned*)(ws + WS_AMAX);
  float*     den   = (float*)(ws + WS_DEN);
  float*     alpha = (float*)(ws + WS_ALPHA);
  float*     agg   = (float*)(ws + WS_AGG);
  float*     out   = (float*)d_out;

  k_swizzle   <<<160, 256, 0, stream>>>(Wl, Wr, Wf, wlr, wfz);
  k_gemm_xlr  <<<dim3(3125, 2), 256, 0, stream>>>(X, bl, br, wlr, XW);
  k_init      <<<25000, 256, 0, stream>>>(bias_g, agg, den, amax);
  k_edge_score<<<53125, 256, 0, stream>>>(E, XW, att, alpha, amax);
  k_edge_expsum<<<(ET * NH + 255) / 256, 256, 0, stream>>>(E, alpha, amax, den);
  k_aggregate <<<(ET + 7) / 8, 256, 0, stream>>>(E, XW, alpha, den, agg);
  k_final_ln  <<<3125, 128, 0, stream>>>(agg, wfz, bf, gamma, beta, out);

  // tuple pass-through outputs: (H_out, E, attr) flat-concatenated
  hipMemcpyAsync(out + (size_t)NNODES * DOUT, E,
                 (size_t)2 * NEDGES * sizeof(int), hipMemcpyDeviceToDevice, stream);
  hipMemcpyAsync(out + (size_t)NNODES * DOUT + 2 * NEDGES, attr,
                 (size_t)NEDGES * sizeof(float), hipMemcpyDeviceToDevice, stream);
}
